// CommonAttention_4183298146333
// MI455X (gfx1250) — matrix operand order repaired and checked by reading, not compiled, not run on hardware
//
#include <hip/hip_runtime.h>

typedef __attribute__((ext_vector_type(16))) _Float16 v16h;
typedef __attribute__((ext_vector_type(8)))  _Float16 v8h;
typedef __attribute__((ext_vector_type(8)))  float    v8f;

#define NB   2
#define NH   12
#define NL   1024
#define NT   1024
#define ND   64
#define NCH  36      // H * FEA real channels; channel 36 = virtual "ones" (bias)
#define TKC  64      // T-chunk for online softmax
#define DIM  768
#define CHSTRIDE ((size_t)NL * NT)   // stride between channels of one score tensor

// ---------------------------------------------------------------------------
// Prep kernels (one-shot, L2-resident data): V -> f16 transposed [b,h,d,t],
// proj_w -> f16. Removes all f32->f16 conversion work from the hot kernels
// and makes every WMMA B-operand two contiguous 16B loads.
// ---------------------------------------------------------------------------
__global__ __launch_bounds__(256) void prep_v_kernel(
    const float* __restrict__ v, _Float16* __restrict__ Vt)
{
  const int idx = blockIdx.x * 256 + threadIdx.x;      // NB*NH*ND*NT exact
  const int t  = idx & (NT - 1);
  const int d  = (idx >> 10) & (ND - 1);
  const int bh = idx >> 16;
  Vt[((size_t)bh * ND + d) * NT + t] = (_Float16)v[((size_t)bh * NT + t) * ND + d];
}

__global__ __launch_bounds__(256) void prep_w_kernel(
    const float* __restrict__ W, _Float16* __restrict__ Wh)
{
  const int idx = blockIdx.x * 256 + threadIdx.x;      // DIM*DIM exact
  Wh[idx] = (_Float16)W[idx];
}

// ---------------------------------------------------------------------------
// Kernel 1: WMMA channel fusion + online (flash-style) softmax + WMMA P@V.
// Grid: NB * NL/16 = 128 blocks; 384 threads = 12 waves, wave w == head w.
// Scores (302 MB) stream through exactly once; next chunk is prefetched
// (global_prefetch_b8) while the current one runs softmax + WMMA.
// ---------------------------------------------------------------------------
__global__ __launch_bounds__(384) void fused_attn_kernel(
    const float* __restrict__ s0, const float* __restrict__ s1,
    const float* __restrict__ s2, const _Float16* __restrict__ Vt,
    const float* __restrict__ fw, const float* __restrict__ fb,
    _Float16* __restrict__ X)
{
  __shared__ float wsh[NH][NCH];
  __shared__ float bsh[NH];
  __shared__ __align__(16) float fused[NH][16][TKC];   // 48 KB

  const int tid = threadIdx.x;
  const int b   = blockIdx.x >> 6;          // 64 l-tiles per batch
  const int l0  = (blockIdx.x & 63) << 4;

  for (int i = tid; i < NH * NCH; i += 384) wsh[i / NCH][i % NCH] = fw[i];
  if (tid < NH) bsh[tid] = fb[tid];
  __syncthreads();

  const int wave = tid >> 5;                // head index
  const int lane = tid & 31;
  const int mrow = lane & 15;               // WMMA A-row / stat row
  const int n16  = lane & 15;               // WMMA B/C column
  const int half = lane >> 4;

  // ---- A operand of the fusion GEMM: rows = output head o (12 used of 16),
  //      K = channel c (0..35 real, 36 = bias channel, 37..63 zero).
  //      Built once, lives in registers for the whole kernel.
  v16h AW0, AW1;
  #pragma unroll
  for (int e = 0; e < 16; ++e) {
    const int K = half * 16 + e;                           // A-layout K for this e
    AW0[e] = (mrow < NH) ? (_Float16)wsh[mrow][K] : (_Float16)0.f;   // c = K (<32)
    float wv = 0.f;                                        // c = 32 + K
    if (half == 0 && mrow < NH) {
      if (e < 4)       wv = wsh[mrow][32 + e];             // c = 32..35
      else if (e == 4) wv = bsh[mrow];                     // c = 36: bias * ones
    }
    AW1[e] = (_Float16)wv;
  }

  float m_run = -3.4e38f;                   // running row max (row = mrow)
  float s_run = 0.0f;                       // running row sum-of-exp
  const v8f zero = {0.f, 0.f, 0.f, 0.f, 0.f, 0.f, 0.f, 0.f};
  v8f acc[4];                               // x[16][64] in C-layout, 4 d-tiles
  acc[0] = zero; acc[1] = zero; acc[2] = zero; acc[3] = zero;

  const _Float16* vtbase = Vt + ((size_t)(b * NH + wave) * ND) * NT;
  const size_t bbase = (size_t)b * NH * CHSTRIDE;

  for (int ic = 0; ic < NT / TKC; ++ic) {
    const int t0 = ic * TKC;

    // ---- fusion GEMM: 64 column-tiles (16 l rows x 4 t-subtiles), RR waves
    for (int tile = wave; tile < 16 * (TKC / 16); tile += NH) {
      const int l    = tile >> 2;
      const int tsub = tile & 3;
      const int tcol = t0 + tsub * 16 + n16;          // this lane's t column
      const size_t off0 = bbase + (size_t)(l0 + l) * NT + tcol;

      // B k-step 0: channels c = half*16 + e  (0..31)
      v16h B0;
      #pragma unroll
      for (int e = 0; e < 16; ++e) {
        // half==0: c=e      -> tensor s0 (e<12) else s1, head e or e-12
        // half==1: c=16+e   -> tensor s1 (e<8)  else s2, head 4+e or e-8
        const float* pl = (e < 12) ? s0 : s1;
        const float* ph = (e < 8)  ? s1 : s2;
        const int hl = (e < 12) ? e : e - 12;
        const int hh = (e < 8) ? 4 + e : e - 8;
        const float* sp = half ? ph : pl;
        const size_t off = off0 + (size_t)(half ? hh : hl) * CHSTRIDE;
        B0[e] = (_Float16)sp[off];
      }
      // B k-step 1: channels c = 32 + half*16 + e; real only for half==0,e<5
      v16h B1;
      #pragma unroll
      for (int e = 0; e < 16; ++e) B1[e] = (_Float16)0.f;
      if (half == 0) {
        #pragma unroll
        for (int e = 0; e < 4; ++e)
          B1[e] = (_Float16)s2[off0 + (size_t)(8 + e) * CHSTRIDE]; // c=32..35
        B1[4] = (_Float16)1.0f;                                    // ones (bias)
      }

      v8f cf = zero;
      cf = __builtin_amdgcn_wmma_f32_16x16x32_f16(false, AW0, false, B0,
                                                  (short)0, cf, false, false);
      cf = __builtin_amdgcn_wmma_f32_16x16x32_f16(false, AW1, false, B1,
                                                  (short)0, cf, false, false);

      // scatter C rows (o = r + 8*half; only o < 12 kept) into fused LDS
      const int col = tsub * 16 + n16;
      #pragma unroll
      for (int r = 0; r < 4; ++r) fused[r + 8 * half][l][col] = cf[r];
      if (half == 0) {
        #pragma unroll
        for (int r = 4; r < 8; ++r) fused[r][l][col] = cf[r];
      }
    }
    __syncthreads();

    // ---- prefetch next chunk's score stream (overlaps HBM with compute) ---
    // 1152 x 128B lines == the full 147 KB of next-chunk scores.
    if (ic + 1 < NT / TKC) {
      const int tnext = t0 + TKC;
      #pragma unroll
      for (int j = 0; j < 3; ++j) {
        const int idx   = tid + j * 384;        // 0..1151
        const int line  = idx & 1;              // which 128B half of a t-row
        const int rowid = idx >> 1;             // 0..575 = (f, hh, l)
        const int f     = rowid / 192;
        const int rem   = rowid - f * 192;
        const int hh    = rem >> 4;
        const int l     = rem & 15;
        const float* sp = (f == 0) ? s0 : (f == 1) ? s1 : s2;
        const float* pa = sp + bbase + (size_t)hh * CHSTRIDE +
                          (size_t)(l0 + l) * NT + tnext + line * 32;
        __builtin_prefetch(pa, 0, 1);
      }
    }

    // ---- online softmax (row = mrow of own head) --------------------------
    const float4* f4 =
        reinterpret_cast<const float4*>(&fused[wave][mrow][half * 32]);
    float cmax = -3.4e38f;
    #pragma unroll
    for (int j = 0; j < 8; ++j) {
      const float4 q = f4[j];
      cmax = fmaxf(cmax, fmaxf(fmaxf(q.x, q.y), fmaxf(q.z, q.w)));
    }
    cmax = fmaxf(cmax, __shfl_xor(cmax, 16, 32));
    const float m_new = fmaxf(m_run, cmax);
    const float scl   = __expf(m_run - m_new);

    float scl8[8];
    #pragma unroll
    for (int r = 0; r < 8; ++r) scl8[r] = __shfl(scl, r + half * 8, 32);
    #pragma unroll
    for (int dt = 0; dt < 4; ++dt) {
      #pragma unroll
      for (int r = 0; r < 8; ++r) acc[dt][r] *= scl8[r];
    }

    // ---- P tile build (exp once: feeds both WMMA-A and the row sum) -------
    float psum = 0.0f;
    #pragma unroll
    for (int ks = 0; ks < TKC / 32; ++ks) {
      const float* fr = &fused[wave][mrow][ks * 32];
      const float4 qa = *reinterpret_cast<const float4*>(fr + half * 8);
      const float4 qb = *reinterpret_cast<const float4*>(fr + half * 8 + 4);
      const float4 qc = *reinterpret_cast<const float4*>(fr + 16 + half * 8);
      const float4 qd = *reinterpret_cast<const float4*>(fr + 16 + half * 8 + 4);
      float p[16];
      p[0] = __expf(qa.x - m_new); p[1] = __expf(qa.y - m_new);
      p[2] = __expf(qa.z - m_new); p[3] = __expf(qa.w - m_new);
      p[4] = __expf(qb.x - m_new); p[5] = __expf(qb.y - m_new);
      p[6] = __expf(qb.z - m_new); p[7] = __expf(qb.w - m_new);
      p[8]  = __expf(qc.x - m_new); p[9]  = __expf(qc.y - m_new);
      p[10] = __expf(qc.z - m_new); p[11] = __expf(qc.w - m_new);
      p[12] = __expf(qd.x - m_new); p[13] = __expf(qd.y - m_new);
      p[14] = __expf(qd.z - m_new); p[15] = __expf(qd.w - m_new);
      v16h A;
      #pragma unroll
      for (int e = 0; e < 16; ++e) { psum += p[e]; A[e] = (_Float16)p[e]; }

      // P @ V: B from pre-transposed f16 Vt -> two contiguous v8h loads
      #pragma unroll
      for (int dt = 0; dt < 4; ++dt) {
        const _Float16* vb = vtbase + (size_t)(dt * 16 + n16) * NT +
                             t0 + ks * 32 + half * 8;
        const v8h b0 = *reinterpret_cast<const v8h*>(vb);
        const v8h b1 = *reinterpret_cast<const v8h*>(vb + 16);
        v16h Bm;
        #pragma unroll
        for (int e = 0; e < 8; ++e) { Bm[e] = b0[e]; Bm[8 + e] = b1[e]; }
        acc[dt] = __builtin_amdgcn_wmma_f32_16x16x32_f16(
            false, A, false, Bm, (short)0, acc[dt], false, false);
      }
    }
    psum += __shfl_xor(psum, 16, 32);
    s_run = s_run * scl + psum;
    m_run = m_new;
    __syncthreads();
  }

  // ---- finalize: divide by row sums, emit merged-head f16 activations -----
  const float invs = 1.0f / s_run;
  float inv8[8];
  #pragma unroll
  for (int r = 0; r < 8; ++r) inv8[r] = __shfl(invs, r + half * 8, 32);
  #pragma unroll
  for (int dt = 0; dt < 4; ++dt) {
    #pragma unroll
    for (int r = 0; r < 8; ++r) {
      const int row = l0 + r + half * 8;
      X[((size_t)b * NL + row) * DIM + wave * ND + dt * 16 + n16] =
          (_Float16)(acc[dt][r] * inv8[r]);
    }
  }
}

// ---------------------------------------------------------------------------
// Kernel 2: out[2048,768] = X_f16 @ Wh^T + bias, WMMA 16x16x32 f16.
// One 16x64 C-strip per wave: A fragment loaded once, reused by 4 WMMAs.
// 128 m-tiles x 12 n-blocks = 1536 wave-units -> 192 blocks of 8 waves.
// ---------------------------------------------------------------------------
__global__ __launch_bounds__(256) void proj_kernel(
    const _Float16* __restrict__ X, const _Float16* __restrict__ Wh,
    const float* __restrict__ bias, float* __restrict__ out)
{
  const int lane = threadIdx.x & 31;
  const int wave = threadIdx.x >> 5;
  const int n16  = lane & 15;     // A row & B col for this lane
  const int half = lane >> 4;

  const int u  = blockIdx.x * 8 + wave;
  const int mt = u / (DIM / 64);
  const int nb = u % (DIM / 64);
  const int m0 = mt * 16;
  const int n0 = nb * 64;

  const v8f zero = {0.f, 0.f, 0.f, 0.f, 0.f, 0.f, 0.f, 0.f};
  v8f c[4];
  c[0] = zero; c[1] = zero; c[2] = zero; c[3] = zero;

  for (int k0 = 0; k0 < DIM; k0 += 32) {
    // A: two contiguous 16B f16 loads (K = half*8+e and 16+half*8+e)
    const _Float16* xa = X + (size_t)(m0 + n16) * DIM + k0 + half * 8;
    const v8h a0 = *reinterpret_cast<const v8h*>(xa);
    const v8h a1 = *reinterpret_cast<const v8h*>(xa + 16);
    v16h A;
    #pragma unroll
    for (int e = 0; e < 8; ++e) { A[e] = a0[e]; A[8 + e] = a1[e]; }

    #pragma unroll
    for (int dt = 0; dt < 4; ++dt) {
      // B[k][n] = Wh[n][k]: K = half*8+e and 16+half*8+e -> two v8h loads
      const _Float16* wb =
          Wh + (size_t)(n0 + dt * 16 + n16) * DIM + k0 + half * 8;
      const v8h b0 = *reinterpret_cast<const v8h*>(wb);
      const v8h b1 = *reinterpret_cast<const v8h*>(wb + 16);
      v16h Bm;
      #pragma unroll
      for (int e = 0; e < 8; ++e) { Bm[e] = b0[e]; Bm[8 + e] = b1[e]; }
      c[dt] = __builtin_amdgcn_wmma_f32_16x16x32_f16(
          false, A, false, Bm, (short)0, c[dt], false, false);
    }
  }

  #pragma unroll
  for (int dt = 0; dt < 4; ++dt) {
    const float bn = bias[n0 + dt * 16 + n16];
    #pragma unroll
    for (int r = 0; r < 8; ++r)
      out[(size_t)(m0 + r + half * 8) * DIM + n0 + dt * 16 + n16] =
          c[dt][r] + bn;
  }
}

// ---------------------------------------------------------------------------
extern "C" void kernel_launch(void* const* d_in, const int* in_sizes, int n_in,
                              void* d_out, int out_size, void* d_ws, size_t ws_size,
                              hipStream_t stream) {
  const float* s0 = (const float*)d_in[0];
  const float* s1 = (const float*)d_in[1];
  const float* s2 = (const float*)d_in[2];
  const float* vg = (const float*)d_in[3];
  const float* fw = (const float*)d_in[4];
  const float* fb = (const float*)d_in[5];
  const float* pw = (const float*)d_in[6];
  const float* pb = (const float*)d_in[7];

  // workspace layout (f16): X[2048*768] | Vt[2*12*64*1024] | Wh[768*768]
  _Float16* X  = (_Float16*)d_ws;
  _Float16* Vt = X  + (size_t)NB * NL * DIM;
  _Float16* Wh = Vt + (size_t)NB * NH * ND * NT;
  float* out   = (float*)d_out;

  prep_v_kernel<<<(NB * NH * ND * NT) / 256, 256, 0, stream>>>(vg, Vt);
  prep_w_kernel<<<(DIM * DIM) / 256, 256, 0, stream>>>(pw, Wh);

  fused_attn_kernel<<<NB * (NL / 16), 384, 0, stream>>>(s0, s1, s2, Vt, fw, fb, X);

  const int waveUnits = (NB * NL / 16) * (DIM / 64);   // 128 * 12 = 1536
  proj_kernel<<<waveUnits / 8, 256, 0, stream>>>(X, Wh, pb, out);
}
